// LabelingLayer_65292092834075
// MI455X (gfx1250) — matrix operand order repaired and checked by reading, not compiled, not run on hardware
//
#include <hip/hip_runtime.h>

// CDNA5 / gfx1250 types
typedef __attribute__((ext_vector_type(16))) _Float16 v16h;
typedef __attribute__((ext_vector_type(8)))  float    v8f;

#define BATCH 128
#define SEQL  1024
#define HID   1024
#define NTAG  9
#define TPAD  16   // padded tag dim for emissions workspace
#define RTILE 2    // row-tiles (of 16 rows) per wave

// pack 16 f32 (4x float4) -> v16h fragment
__device__ __forceinline__ v16h cvt16(float4 a, float4 b, float4 c, float4 d) {
    v16h r;
    r[0]=(_Float16)a.x;  r[1]=(_Float16)a.y;  r[2]=(_Float16)a.z;  r[3]=(_Float16)a.w;
    r[4]=(_Float16)b.x;  r[5]=(_Float16)b.y;  r[6]=(_Float16)b.z;  r[7]=(_Float16)b.w;
    r[8]=(_Float16)c.x;  r[9]=(_Float16)c.y;  r[10]=(_Float16)c.z; r[11]=(_Float16)c.w;
    r[12]=(_Float16)d.x; r[13]=(_Float16)d.y; r[14]=(_Float16)d.z; r[15]=(_Float16)d.w;
    return r;
}

// ---------------------------------------------------------------------------
// Kernel 1: emissions[B*L, 16] = enc[B*L, 1024] x W^T[1024, 9] + b
// One wave computes TWO 16x16 output tiles sharing one B fragment per K-step.
// Double-buffered staging: loads for K-step kb+32 are issued before the
// convert+WMMA of K-step kb, keeping ~12 b128 loads in flight per wave.
// A-fragment layout (16-bit A 16x32): lane m(0-15)/m+16 holds row m;
//   lanes<16: K = kb+0..7 and kb+16..23 ; lanes>=16: K = kb+8..15 and kb+24..31.
// B-fragment layout (16-bit B 32x16): lane n(0-15)/n+16 holds col n;
//   lanes<16: K = kb+0..7 and kb+16..23 ; lanes>=16: K = kb+8..15 and kb+24..31.
// C/D layout: lane group selects N, VGPR v selects M = v + 8*(lane>=16).
// ---------------------------------------------------------------------------
__global__ __launch_bounds__(32) void emis_gemm_wmma(
    const float* __restrict__ A,     // [B*L, HID]
    const float* __restrict__ W,     // [NTAG, HID]
    const float* __restrict__ bias,  // [NTAG]
    float* __restrict__ emis)        // [B*L, TPAD]
{
    const int lane = threadIdx.x;
    const int hi   = lane >> 4;          // 0 or 1
    const int m    = lane & 15;          // row within tile / col index n

    const int aKo = hi << 3;             // A half-offset: 0 or 8
    const int bKo = hi << 3;             // B half-offset: 0 or 8

    const long row0 = (long)blockIdx.x * (16 * RTILE) + m;
    const float* __restrict__ Ar0 = A + row0 * HID;           // tile 0 row
    const float* __restrict__ Ar1 = Ar0 + 16L * HID;          // tile 1 row

    const int n  = m;
    const int wn = (n < NTAG) ? n : (NTAG - 1);   // clamp: cols 9..15 discarded
    const float* __restrict__ Wrow = W + wn * HID;

    v8f acc0 = {}, acc1 = {};

    // ---- preload K-step 0 staging ----
    float4 w0  = *(const float4*)(Wrow + bKo);
    float4 w1  = *(const float4*)(Wrow + bKo + 4);
    float4 w2  = *(const float4*)(Wrow + bKo + 16);
    float4 w3  = *(const float4*)(Wrow + bKo + 20);
    float4 a00 = *(const float4*)(Ar0 + aKo);
    float4 a01 = *(const float4*)(Ar0 + aKo + 4);
    float4 a02 = *(const float4*)(Ar0 + 16 + aKo);
    float4 a03 = *(const float4*)(Ar0 + 16 + aKo + 4);
    float4 a10 = *(const float4*)(Ar1 + aKo);
    float4 a11 = *(const float4*)(Ar1 + aKo + 4);
    float4 a12 = *(const float4*)(Ar1 + 16 + aKo);
    float4 a13 = *(const float4*)(Ar1 + 16 + aKo + 4);

    for (int kb = 0; kb < HID; kb += 32) {
        // speculative prefetch ~8 iterations ahead (dropped past end of row/array)
        __builtin_prefetch(Ar0 + kb + 256 + aKo, 0, 0);
        __builtin_prefetch(Ar1 + kb + 256 + aKo, 0, 0);

        // ---- issue next K-step's 12 loads before using current staging ----
        const int kn = (kb + 32 < HID) ? (kb + 32) : kb;  // last iter: harmless reload
        float4 nw0  = *(const float4*)(Wrow + kn + bKo);
        float4 nw1  = *(const float4*)(Wrow + kn + bKo + 4);
        float4 nw2  = *(const float4*)(Wrow + kn + bKo + 16);
        float4 nw3  = *(const float4*)(Wrow + kn + bKo + 20);
        float4 na00 = *(const float4*)(Ar0 + kn + aKo);
        float4 na01 = *(const float4*)(Ar0 + kn + aKo + 4);
        float4 na02 = *(const float4*)(Ar0 + kn + 16 + aKo);
        float4 na03 = *(const float4*)(Ar0 + kn + 16 + aKo + 4);
        float4 na10 = *(const float4*)(Ar1 + kn + aKo);
        float4 na11 = *(const float4*)(Ar1 + kn + aKo + 4);
        float4 na12 = *(const float4*)(Ar1 + kn + 16 + aKo);
        float4 na13 = *(const float4*)(Ar1 + kn + 16 + aKo + 4);

        // ---- convert current staging and multiply ----
        const v16h bv  = cvt16(w0, w1, w2, w3);
        const v16h av0 = cvt16(a00, a01, a02, a03);
        const v16h av1 = cvt16(a10, a11, a12, a13);

        // (neg_a, A, neg_b, B, c_mod, C, reuse_a, reuse_b)
        acc0 = __builtin_amdgcn_wmma_f32_16x16x32_f16(
            false, av0, false, bv, (short)0, acc0, false, false);
        acc1 = __builtin_amdgcn_wmma_f32_16x16x32_f16(
            false, av1, false, bv, (short)0, acc1, false, false);

        // ---- rotate buffers ----
        w0=nw0; w1=nw1; w2=nw2; w3=nw3;
        a00=na00; a01=na01; a02=na02; a03=na03;
        a10=na10; a11=na11; a12=na12; a13=na13;
    }

    const float bb = (n < NTAG) ? bias[n] : 0.f;
    float* __restrict__ out = emis + (long)blockIdx.x * (16 * RTILE) * TPAD;
#pragma unroll
    for (int v = 0; v < 8; ++v) {
        const int mm = v + (hi << 3);
        out[mm * TPAD + n]              = acc0[v] + bb;
        out[(16 + mm) * TPAD + n]       = acc1[v] + bb;
    }
}

// ---------------------------------------------------------------------------
// Kernel 2: Viterbi decode. One wave per batch; lane = tag (lanes 9..31 idle-ish).
// Score exchange via __shfl (ds_bpermute, wave32). Backpointers in LDS so the
// sequential backtrack is LDS-latency bound, not an HBM pointer chase.
// ---------------------------------------------------------------------------
__global__ __launch_bounds__(32) void viterbi_kernel(
    const float* __restrict__ emis,    // [B*L, TPAD]
    const int*   __restrict__ mask,    // [B, L] 0/1
    const float* __restrict__ startT,  // [NTAG]
    const float* __restrict__ endT,    // [NTAG]
    const float* __restrict__ trans,   // [NTAG, NTAG]  trans[i*9+j]
    float* __restrict__ out_tags,      // [B, L] (tags as float)
    float* __restrict__ out_score)     // [B]
{
    __shared__ unsigned char hist[(SEQL - 1) * TPAD];  // ~16 KB of 320 KB/WGP

    const int b    = blockIdx.x;
    const int lane = threadIdx.x;
    const int j    = lane;
    const int jj   = (j < NTAG) ? j : 0;
    const float NEG = -3.0e38f;

    // transitions column j (incoming edges i -> j), resident in registers
    float trj[NTAG];
#pragma unroll
    for (int i = 0; i < NTAG; ++i) trj[i] = trans[i * NTAG + jj];

    const long ebase = (long)b * SEQL * TPAD;
    float s = (j < NTAG) ? (startT[jj] + emis[ebase + jj]) : NEG;

    for (int l = 1; l < SEQL; ++l) {
        const float e  = emis[ebase + l * TPAD + jj];
        const int   mk = mask[b * SEQL + l];           // uniform across wave

        float best = NEG;
        int   bi   = 0;
#pragma unroll
        for (int i = 0; i < NTAG; ++i) {
            const float si = __shfl(s, i, 32);         // ds_bpermute_b32
            const float v  = si + trj[i];
            if (v > best) { best = v; bi = i; }
        }
        float ns = best + e;
        if (!mk) { ns = s; bi = jj; }                  // frozen step: identity ptr
        if (j < NTAG) hist[(l - 1) * TPAD + j] = (unsigned char)bi;
        s = ns;
    }
    __syncthreads();

    // final = score + end_transitions ; argmax over tags (first-max tiebreak)
    float fbest = NEG;
    int   btag  = 0;
#pragma unroll
    for (int i = 0; i < NTAG; ++i) {
        const float fi = __shfl(s, i, 32) + endT[i];
        if (fi > fbest) { fbest = fi; btag = i; }
    }

    if (lane == 0) {
        out_score[b] = fbest;
        int tag = btag;
        for (int l = SEQL - 1; l >= 1; --l) {
            const int mk = mask[b * SEQL + l];
            out_tags[b * SEQL + l] = mk ? (float)tag : 0.f;
            tag = hist[(l - 1) * TPAD + tag];
        }
        out_tags[b * SEQL] = (float)tag;               // mask[:,0] is all-true
    }
}

// ---------------------------------------------------------------------------
extern "C" void kernel_launch(void* const* d_in, const int* in_sizes, int n_in,
                              void* d_out, int out_size, void* d_ws, size_t ws_size,
                              hipStream_t stream) {
    (void)in_sizes; (void)n_in; (void)out_size; (void)ws_size;

    const float* enc    = (const float*)d_in[0];  // [128,1024,1024] f32
    const int*   mask   = (const int*)  d_in[1];  // [128,1024]
    const float* W      = (const float*)d_in[2];  // [9,1024]
    const float* bias   = (const float*)d_in[3];  // [9]
    const float* startT = (const float*)d_in[4];  // [9]
    const float* endT   = (const float*)d_in[5];  // [9]
    const float* trans  = (const float*)d_in[6];  // [9,9]

    float* out_tags  = (float*)d_out;                 // [128*1024]
    float* out_score = (float*)d_out + BATCH * SEQL;  // [128]
    float* emis      = (float*)d_ws;                  // [128*1024, 16] = 8 MB

    const int nblocks = (BATCH * SEQL) / (16 * RTILE);  // 4096 waves x 2 tiles
    emis_gemm_wmma<<<nblocks, 32, 0, stream>>>(enc, W, bias, emis);
    viterbi_kernel<<<BATCH, 32, 0, stream>>>(emis, mask, startT, endT, trans,
                                             out_tags, out_score);
}
